// EncoderBlock_62740882260693
// MI455X (gfx1250) — matrix operand order repaired and checked by reading, not compiled, not run on hardware
//
#include <hip/hip_runtime.h>
#include <hip/hip_bf16.h>
#include <math.h>

// ---------------------------------------------------------------------------
// Transformer encoder block (pre-LN) for MI455X / gfx1250, bf16 WMMA path.
// B=2, S=2048, E=1024, H=16, DH=64, FF=4096. All GEMMs use
// v_wmma_f32_16x16x32_bf16 with f32 accumulation. Tile staging uses
// GLOBAL_LOAD_ASYNC_TO_LDS_B128 (ASYNCcnt), double-buffered so the DMA for
// tile i+1 is in flight while tile i's WMMAs execute.
// ---------------------------------------------------------------------------

typedef __bf16 bf16;
typedef __attribute__((ext_vector_type(16))) __bf16 v16bf;
typedef __attribute__((ext_vector_type(8)))  __bf16 v8bf;
typedef __attribute__((ext_vector_type(8)))  float  v8f;
typedef __attribute__((ext_vector_type(4)))  int    v4i;

#define E_DIM 1024
#define H_NUM 16
#define DHEAD 64
#define FF_DIM 4096
#define BSROWS 4096   // B*S
#define SEQ 2048
#define E3 3072       // 3*E (fused QKV width)

static __device__ __forceinline__ bf16 f2b(float x) { return (bf16)x; }

static __device__ __forceinline__ v8f zero8() {
  v8f z;
#pragma unroll
  for (int i = 0; i < 8; i++) z[i] = 0.f;
  return z;
}

// Build a 16-element bf16 fragment from two contiguous 8-element chunks.
// Element order matches the ISA VGPR layout: elems 0..7 -> VGPRs 0..3,
// elems 8..15 -> VGPRs 4..7 (2 bf16 per VGPR).
static __device__ __forceinline__ v16bf make_frag(const bf16* p0, const bf16* p1) {
  v8bf lo = *(const v8bf*)p0;
  v8bf hi = *(const v8bf*)p1;
  v16bf r;
#pragma unroll
  for (int i = 0; i < 8; i++) { r[i] = lo[i]; r[8 + i] = hi[i]; }
  return r;
}

static __device__ __forceinline__ v8f wmma_bf(v16bf a, v16bf b, v8f c) {
  return __builtin_amdgcn_wmma_f32_16x16x32_bf16(
      /*neg_a=*/false, a, /*neg_b=*/false, b,
      /*c_mod=*/(short)0, c, /*reuse_a=*/false, /*reuse_b=*/false);
}

// Async 16-byte global -> LDS copy (tracked with ASYNCcnt on gfx1250).
static __device__ __forceinline__ void async_ld_b128(const bf16* gptr, bf16* lptr) {
#if defined(__gfx1250__)
#if __has_builtin(__builtin_amdgcn_global_load_async_to_lds_b128)
  __builtin_amdgcn_global_load_async_to_lds_b128(
      (__attribute__((address_space(1))) v4i*)gptr,
      (__attribute__((address_space(3))) v4i*)lptr, 0, 0);
#else
  unsigned int       l = (unsigned int)(unsigned long long)lptr;
  unsigned long long g = (unsigned long long)gptr;
  asm volatile("global_load_async_to_lds_b128 %0, %1, off"
               :: "v"(l), "v"(g) : "memory");
#endif
#else
  *(v8bf*)lptr = *(const v8bf*)gptr;  // host-pass placeholder
#endif
}

template <int N>
static __device__ __forceinline__ void wait_asyncN() {
#if defined(__gfx1250__)
#if __has_builtin(__builtin_amdgcn_s_wait_asynccnt)
  __builtin_amdgcn_s_wait_asynccnt(N);
#else
  asm volatile("s_wait_asynccnt %0" :: "i"(N) : "memory");
#endif
#endif
}

// ---------------------------------------------------------------------------
// Weight packing / conversion kernels
// ---------------------------------------------------------------------------

__global__ void cvt_kernel(const float* __restrict__ in, bf16* __restrict__ out, int n) {
  int i = blockIdx.x * blockDim.x + threadIdx.x;
  if (i < n) out[i] = f2b(in[i]);
}

// wq/wk/wv: [H, E, DH] -> fused Wqkv bf16 [E, 3E], col = which*E + h*DH + d
__global__ void pack_qkv_w(const float* __restrict__ wq, const float* __restrict__ wk,
                           const float* __restrict__ wv, bf16* __restrict__ out) {
  int i = blockIdx.x * blockDim.x + threadIdx.x;       // 3 * 1M
  if (i >= 3 * E_DIM * E_DIM) return;
  int which = i >> 20;
  int rem = i & 0xFFFFF;                               // h*E*DH + e*DH + d
  const float* src = (which == 0) ? wq : ((which == 1) ? wk : wv);
  float v = src[rem];
  int h = rem / (E_DIM * DHEAD);
  int e = (rem / DHEAD) % E_DIM;
  int d = rem % DHEAD;
  out[(size_t)e * E3 + which * E_DIM + h * DHEAD + d] = f2b(v);
}

__global__ void pack_qkv_b(const float* __restrict__ bq, const float* __restrict__ bk,
                           const float* __restrict__ bv, float* __restrict__ out) {
  int i = blockIdx.x * blockDim.x + threadIdx.x;       // 3072
  if (i >= E3) return;
  const float* s = (i < E_DIM) ? bq : ((i < 2 * E_DIM) ? bk : bv);
  out[i] = s[i & (E_DIM - 1)];
}

// ---------------------------------------------------------------------------
// LayerNorm: one row (E=1024) per block, output bf16
// ---------------------------------------------------------------------------

__global__ __launch_bounds__(256) void ln_kernel(const float* __restrict__ x,
                                                 const float* __restrict__ gam,
                                                 const float* __restrict__ bet,
                                                 bf16* __restrict__ out) {
  __shared__ float s1[256], s2[256];
  int row = blockIdx.x;
  const float* xr = x + (size_t)row * E_DIM;
  float v[4], sum = 0.f, sq = 0.f;
#pragma unroll
  for (int i = 0; i < 4; i++) {
    v[i] = xr[threadIdx.x + i * 256];
    sum += v[i];
    sq  += v[i] * v[i];
  }
  s1[threadIdx.x] = sum;
  s2[threadIdx.x] = sq;
  __syncthreads();
  for (int o = 128; o > 0; o >>= 1) {
    if (threadIdx.x < o) {
      s1[threadIdx.x] += s1[threadIdx.x + o];
      s2[threadIdx.x] += s2[threadIdx.x + o];
    }
    __syncthreads();
  }
  float mu  = s1[0] * (1.f / E_DIM);
  float var = s2[0] * (1.f / E_DIM) - mu * mu;
  float inv = rsqrtf(var + 1e-5f);
  bf16* orow = out + (size_t)row * E_DIM;
#pragma unroll
  for (int i = 0; i < 4; i++) {
    int c = threadIdx.x + i * 256;
    orow[c] = f2b((v[i] - mu) * inv * gam[c] + bet[c]);
  }
}

// ---------------------------------------------------------------------------
// Tiled bf16 WMMA GEMM: C[M,N] = A[M,K] * B[K,N] + bias, fused epilogue.
//   EPI 0: out bf16, columns < E scaled by 0.125 (fused QKV, Q pre-scale)
//   EPI 1: out f32  = resid + acc + bias (residual stream)
//   EPI 2: out bf16 = exact GELU(acc + bias)
// Block 256 threads = 8 waves, block tile 128x128, K-step 32, double-buffered.
// Wave (wm in 0..1, wn in 0..3) computes 64x32 = 4x2 WMMA tiles.
// A tile staged via async global->LDS DMA; B tile transposed through VGPRs
// with the ds_store deferred until after the WMMAs (hides global latency).
// ---------------------------------------------------------------------------

template <int EPI>
__global__ __launch_bounds__(256) void gemm_bf16_kernel(
    const bf16* __restrict__ A, const bf16* __restrict__ B,
    const float* __restrict__ bias, const float* __restrict__ resid,
    float* __restrict__ outf, bf16* __restrict__ outb,
    int M, int N, int K) {
  __shared__ bf16 As[2][128 * 32];   // [row][k]  row-major
  __shared__ bf16 BsT[2][128 * 32];  // [col][k]  transposed for B-frag reads

  int tid  = threadIdx.x;
  int lane = tid & 31, wid = tid >> 5;
  int g = lane >> 4, ln = lane & 15;
  int wm = wid >> 2, wn = wid & 3;
  int m0 = blockIdx.y * 128, n0 = blockIdx.x * 128;

  v8f acc[4][2];
#pragma unroll
  for (int i = 0; i < 4; i++)
#pragma unroll
    for (int j = 0; j < 2; j++) acc[i][j] = zero8();

  int arow = tid >> 1, acol = (tid & 1) * 16;   // A: 128 rows x 32 k
  int brow = tid >> 3, bcol = (tid & 7) * 16;   // B: 32 k x 128 cols

  const bf16* ga = A + (size_t)(m0 + arow) * K + acol;
  const bf16* gb = B + (size_t)brow * N + n0 + bcol;
  const size_t bstep = (size_t)32 * N;

  // ---- prologue: stage tile 0 ----
  async_ld_b128(ga,     &As[0][arow * 32 + acol]);
  async_ld_b128(ga + 8, &As[0][arow * 32 + acol + 8]);
  ga += 32;
  {
    v8bf b0 = *(const v8bf*)gb;
    v8bf b1 = *(const v8bf*)(gb + 8);
    gb += bstep;
#pragma unroll
    for (int i = 0; i < 8; i++) {
      BsT[0][(bcol + i)     * 32 + brow] = b0[i];
      BsT[0][(bcol + 8 + i) * 32 + brow] = b1[i];
    }
  }

  const int steps = K >> 5;
  for (int kt = 0; kt < steps; ++kt) {
    const int b = kt & 1;
    const bool hn = (kt + 1) < steps;
    __syncthreads();                       // prev readers of buf b^1 done

    v8bf nb0, nb1;
    if (hn) {
      async_ld_b128(ga,     &As[b ^ 1][arow * 32 + acol]);
      async_ld_b128(ga + 8, &As[b ^ 1][arow * 32 + acol + 8]);
      nb0 = *(const v8bf*)gb;              // next B tile -> regs (stored later)
      nb1 = *(const v8bf*)(gb + 8);
      __builtin_prefetch(ga + 32, 0, 1);   // tile kt+2: keep L2 warm
      __builtin_prefetch(gb + bstep, 0, 1);
      ga += 32;
      gb += bstep;
      wait_asyncN<2>();                    // tile kt's DMAs done; kt+1 in flight
    } else {
      wait_asyncN<0>();                    // final tile: drain
    }
    __syncthreads();                       // tile kt visible to all waves

    v16bf af[4], bfq[2];
#pragma unroll
    for (int i = 0; i < 4; i++) {
      const bf16* p = &As[b][(wm * 64 + i * 16 + ln) * 32];
      af[i] = make_frag(p + g * 8, p + 16 + g * 8);   // A: K chunks g*8, 16+g*8
    }
#pragma unroll
    for (int j = 0; j < 2; j++) {
      const bf16* p = &BsT[b][(wn * 32 + j * 16 + ln) * 32];
      bfq[j] = make_frag(p + g * 8, p + 16 + g * 8);  // B: K chunks g*8, 16+g*8
    }
#pragma unroll
    for (int i = 0; i < 4; i++)
#pragma unroll
      for (int j = 0; j < 2; j++)
        acc[i][j] = wmma_bf(af[i], bfq[j], acc[i][j]);

    if (hn) {                              // deferred transpose-store of next B
#pragma unroll
      for (int i = 0; i < 8; i++) {
        BsT[b ^ 1][(bcol + i)     * 32 + brow] = nb0[i];
        BsT[b ^ 1][(bcol + 8 + i) * 32 + brow] = nb1[i];
      }
    }
  }

  // Epilogue. C/D layout: VGPR r -> row r + g*8, lanes hold cols 0..15.
#pragma unroll
  for (int i = 0; i < 4; i++) {
#pragma unroll
    for (int j = 0; j < 2; j++) {
      int col = n0 + wn * 32 + j * 16 + ln;
      float bval = bias[col];
#pragma unroll
      for (int r = 0; r < 8; r++) {
        int row = m0 + wm * 64 + i * 16 + g * 8 + r;
        size_t idx = (size_t)row * N + col;
        float v = acc[i][j][r] + bval;
        if constexpr (EPI == 0) {
          outb[idx] = f2b(col < E_DIM ? v * 0.125f : v);
        } else if constexpr (EPI == 1) {
          outf[idx] = resid[idx] + v;
        } else {
          outb[idx] = f2b(0.5f * v * (1.f + erff(v * 0.70710678118654752f)));
        }
      }
    }
  }
}

// ---------------------------------------------------------------------------
// Flash attention, one (b,h) per blockIdx.x, 128 query rows per blockIdx.y.
// 8 waves; each wave owns a 16-row query tile. K tiles staged via async DMA,
// V tiles transposed through VGPRs; both double-buffered like the GEMM.
// Online softmax (scores pre-scaled by 1/sqrt(DH) via Q); P routed through
// per-wave LDS (C-layout -> A-layout). Output bf16 head-concat [B*S, E].
// ---------------------------------------------------------------------------

__global__ __launch_bounds__(256) void attn_kernel(const bf16* __restrict__ qkv,
                                                   bf16* __restrict__ O) {
  __shared__ bf16 Ks[2][32 * 64];    // [t_local][d]
  __shared__ bf16 VT[2][64 * 32];    // [d][t_local] (transposed V)
  __shared__ bf16 Pb[8][16 * 32];    // per-wave P tile [m][t_local]

  int tid  = threadIdx.x;
  int lane = tid & 31, wid = tid >> 5;
  int g = lane >> 4, ln = lane & 15;
  int b = blockIdx.x >> 4, h = blockIdx.x & 15;
  int q0 = blockIdx.y * 128 + wid * 16;

  // Q fragments: 16 rows x 64 d -> two K=32 fragments, direct from global.
  const bf16* qrow = qkv + ((size_t)(b * SEQ + q0 + ln)) * E3 + h * DHEAD;
  v16bf qa[2];
#pragma unroll
  for (int ks = 0; ks < 2; ks++)
    qa[ks] = make_frag(qrow + ks * 32 + g * 8, qrow + ks * 32 + 16 + g * 8);

  v8f o[4];
#pragma unroll
  for (int dt = 0; dt < 4; dt++) o[dt] = zero8();
  float ms[8], ls[8];
#pragma unroll
  for (int r = 0; r < 8; r++) { ms[r] = -1e30f; ls[r] = 0.f; }

  int srow = tid >> 3;             // 0..31 (t_local)
  int scol = (tid & 7) * 8;        // 0..56 step 8 (d)

  const bf16* kg = qkv + ((size_t)(b * SEQ + srow)) * E3 +     E_DIM + h * DHEAD + scol;
  const bf16* vg = qkv + ((size_t)(b * SEQ + srow)) * E3 + 2 * E_DIM + h * DHEAD + scol;
  const size_t step = (size_t)32 * E3;

  // ---- prologue: stage tile 0 ----
  async_ld_b128(kg, &Ks[0][srow * 64 + scol]);
  kg += step;
  {
    v8bf vv = *(const v8bf*)vg;
    vg += step;
#pragma unroll
    for (int i = 0; i < 8; i++) VT[0][(scol + i) * 32 + srow] = vv[i];
  }

  const int steps = SEQ / 32;
  for (int tt = 0; tt < steps; ++tt) {
    const int bb = tt & 1;
    const bool hn = (tt + 1) < steps;
    __syncthreads();                       // prev readers of buf bb^1 done

    v8bf vn;
    if (hn) {
      async_ld_b128(kg, &Ks[bb ^ 1][srow * 64 + scol]);
      vn = *(const v8bf*)vg;               // next V tile -> regs (stored later)
      __builtin_prefetch(kg + step, 0, 1);
      __builtin_prefetch(vg + step, 0, 1);
      kg += step;
      vg += step;
      wait_asyncN<1>();                    // tile tt's K DMA done; tt+1 in flight
    } else {
      wait_asyncN<0>();
    }
    __syncthreads();                       // tile tt visible to all waves

    // scores: S[s, t0+0..31] = Q . K^T  (two 16-col tiles, K-dim = 64)
    v8f sc0 = zero8(), sc1 = zero8();
#pragma unroll
    for (int ks = 0; ks < 2; ks++) {
      const bf16* kb0 = &Ks[bb][(ln)      * 64 + ks * 32 + g * 8];
      const bf16* kb1 = &Ks[bb][(16 + ln) * 64 + ks * 32 + g * 8];
      sc0 = wmma_bf(qa[ks], make_frag(kb0, kb0 + 16), sc0);
      sc1 = wmma_bf(qa[ks], make_frag(kb1, kb1 + 16), sc1);
    }

    // online softmax: row (r,g) spans 16 lanes of this half-wave
    float p0[8], p1[8];
#pragma unroll
    for (int r = 0; r < 8; r++) {
      float mx = fmaxf(sc0[r], sc1[r]);
#pragma unroll
      for (int msk = 1; msk < 16; msk <<= 1) mx = fmaxf(mx, __shfl_xor(mx, msk));
      float mn = fmaxf(ms[r], mx);
      float rescale = __expf(ms[r] - mn);
      ms[r] = mn;
      float a0 = __expf(sc0[r] - mn);
      float a1 = __expf(sc1[r] - mn);
      p0[r] = a0; p1[r] = a1;
      float rs = a0 + a1;
#pragma unroll
      for (int msk = 1; msk < 16; msk <<= 1) rs += __shfl_xor(rs, msk);
      ls[r] = ls[r] * rescale + rs;
#pragma unroll
      for (int dt = 0; dt < 4; dt++) o[dt][r] *= rescale;
    }

    // Convert P from C-layout to A-layout via per-wave LDS scratch.
    bf16* pw = Pb[wid];
#pragma unroll
    for (int r = 0; r < 8; r++) {
      pw[(g * 8 + r) * 32 + ln]      = f2b(p0[r]);
      pw[(g * 8 + r) * 32 + 16 + ln] = f2b(p1[r]);
    }
    const bf16* pr = &pw[ln * 32];
    v16bf pa = make_frag(pr + g * 8, pr + 16 + g * 8);

    // O += P (16x32) x V (32x64): four 16-wide d tiles, single K=32 step.
#pragma unroll
    for (int dt = 0; dt < 4; dt++) {
      const bf16* vb = &VT[bb][(dt * 16 + ln) * 32 + g * 8];
      o[dt] = wmma_bf(pa, make_frag(vb, vb + 16), o[dt]);
    }

    if (hn) {                              // deferred transpose-store of next V
#pragma unroll
      for (int i = 0; i < 8; i++) VT[bb ^ 1][(scol + i) * 32 + srow] = vn[i];
    }
  }

  // normalize and write head-concat output
#pragma unroll
  for (int r = 0; r < 8; r++) {
    float inv = 1.f / ls[r];
    int row = q0 + g * 8 + r;
    bf16* orow = O + ((size_t)(b * SEQ + row)) * E_DIM + h * DHEAD;
#pragma unroll
    for (int dt = 0; dt < 4; dt++) orow[dt * 16 + ln] = f2b(o[dt][r] * inv);
  }
}

// ---------------------------------------------------------------------------
// Host launcher
// ---------------------------------------------------------------------------

extern "C" void kernel_launch(void* const* d_in, const int* in_sizes, int n_in,
                              void* d_out, int out_size, void* d_ws, size_t ws_size,
                              hipStream_t stream) {
  (void)in_sizes; (void)n_in; (void)out_size; (void)ws_size;

  const float* x    = (const float*)d_in[0];
  const float* ln1g = (const float*)d_in[1];
  const float* ln1b = (const float*)d_in[2];
  const float* ln2g = (const float*)d_in[3];
  const float* ln2b = (const float*)d_in[4];
  const float* wq   = (const float*)d_in[5];
  const float* bq   = (const float*)d_in[6];
  const float* wk   = (const float*)d_in[7];
  const float* bk   = (const float*)d_in[8];
  const float* wv   = (const float*)d_in[9];
  const float* bv   = (const float*)d_in[10];
  const float* wo   = (const float*)d_in[11];
  const float* bo   = (const float*)d_in[12];
  const float* w1   = (const float*)d_in[13];
  const float* b1   = (const float*)d_in[14];
  const float* w2   = (const float*)d_in[15];
  const float* b2   = (const float*)d_in[16];

  char* ws = (char*)d_ws;
  size_t off = 0;
  auto alloc = [&](size_t bytes) -> void* {
    void* p = ws + off;
    off += (bytes + 255) & ~(size_t)255;
    return p;
  };
  bf16*  wqkv_b = (bf16*) alloc((size_t)E_DIM * E3 * 2);
  float* bqkv   = (float*)alloc((size_t)E3 * 4);
  bf16*  wo_b   = (bf16*) alloc((size_t)E_DIM * E_DIM * 2);
  bf16*  w1_b   = (bf16*) alloc((size_t)E_DIM * FF_DIM * 2);
  bf16*  w2_b   = (bf16*) alloc((size_t)FF_DIM * E_DIM * 2);
  bf16*  h1     = (bf16*) alloc((size_t)BSROWS * E_DIM * 2);
  bf16*  qkvb   = (bf16*) alloc((size_t)BSROWS * E3 * 2);
  bf16*  ob     = (bf16*) alloc((size_t)BSROWS * E_DIM * 2);
  float* x2     = (float*)alloc((size_t)BSROWS * E_DIM * 4);
  bf16*  h2     = (bf16*) alloc((size_t)BSROWS * E_DIM * 2);
  bf16*  ffb    = (bf16*) alloc((size_t)BSROWS * FF_DIM * 2);

  // --- weight packing (bf16) ---
  pack_qkv_w<<<(3 * E_DIM * E_DIM) / 256, 256, 0, stream>>>(wq, wk, wv, wqkv_b);
  pack_qkv_b<<<E3 / 256, 256, 0, stream>>>(bq, bk, bv, bqkv);
  cvt_kernel<<<(E_DIM * E_DIM) / 256, 256, 0, stream>>>(wo, wo_b, E_DIM * E_DIM);
  cvt_kernel<<<(E_DIM * FF_DIM) / 256, 256, 0, stream>>>(w1, w1_b, E_DIM * FF_DIM);
  cvt_kernel<<<(FF_DIM * E_DIM) / 256, 256, 0, stream>>>(w2, w2_b, FF_DIM * E_DIM);

  // --- attention ---
  ln_kernel<<<BSROWS, 256, 0, stream>>>(x, ln1g, ln1b, h1);
  gemm_bf16_kernel<0><<<dim3(E3 / 128, BSROWS / 128), 256, 0, stream>>>(
      h1, wqkv_b, bqkv, nullptr, nullptr, qkvb, BSROWS, E3, E_DIM);
  attn_kernel<<<dim3(2 * H_NUM, SEQ / 128), 256, 0, stream>>>(qkvb, ob);
  gemm_bf16_kernel<1><<<dim3(E_DIM / 128, BSROWS / 128), 256, 0, stream>>>(
      ob, wo_b, bo, x, x2, nullptr, BSROWS, E_DIM, E_DIM);

  // --- feed-forward ---
  ln_kernel<<<BSROWS, 256, 0, stream>>>(x2, ln2g, ln2b, h2);
  gemm_bf16_kernel<2><<<dim3(FF_DIM / 128, BSROWS / 128), 256, 0, stream>>>(
      h2, w1_b, b1, nullptr, nullptr, ffb, BSROWS, FF_DIM, E_DIM);
  gemm_bf16_kernel<1><<<dim3(E_DIM / 128, BSROWS / 128), 256, 0, stream>>>(
      ffb, w2_b, b2, x2, (float*)d_out, nullptr, BSROWS, E_DIM, FF_DIM);
}
